// ImplicitHopfieldSynapse_52329881535148
// MI455X (gfx1250) — compile-verified, matrix operand order checked by reading, not run on hardware
//
#include <hip/hip_runtime.h>
#include <math.h>

#define MDIM 8192   // B*T
#define KDIM 4096   // reduction dim (n)
#define HDIM 16384  // hidden
#define BM 128
#define BH 128
#define BK 32
#define HB (HDIM / BH)   // 128 h-blocks
#define NT (KDIM / BK)   // 128 k-steps
#define LSTR 40          // LDS row stride in ushorts (32 data + 4dw pad), 80B rows -> 16B aligned

typedef __bf16 bf16x16 __attribute__((ext_vector_type(16)));
typedef float  f32x8   __attribute__((ext_vector_type(8)));
typedef unsigned int u32x4 __attribute__((ext_vector_type(4)));
typedef int i32x8 __attribute__((ext_vector_type(8)));
typedef int i32x4 __attribute__((ext_vector_type(4)));

union FragBF {
    bf16x16 v;
    uint4 q[2];
};

__device__ __forceinline__ unsigned short f2bf(float f) {
    unsigned int u = __float_as_uint(f);
    u += 0x7FFFu + ((u >> 16) & 1u);   // round-to-nearest-even
    return (unsigned short)(u >> 16);
}
__device__ __forceinline__ unsigned int pack_bf2(float a, float b) {
    return (unsigned int)f2bf(a) | ((unsigned int)f2bf(b) << 16);
}

// ---------------- Kernel 1a: partial column sum-of-squares of W ----------------
__global__ void norm_partial(const float* __restrict__ W, float* __restrict__ psq) {
    int h  = blockIdx.x * blockDim.x + threadIdx.x;
    int n0 = blockIdx.y * (KDIM / 8);
    const float* p = W + (size_t)n0 * HDIM + h;
    float a0 = 0.f, a1 = 0.f, a2 = 0.f, a3 = 0.f;
#pragma unroll 4
    for (int n = 0; n < KDIM / 8; n += 4) {
        float w0 = p[0 * HDIM], w1 = p[1 * HDIM], w2 = p[2 * HDIM], w3 = p[3 * HDIM];
        a0 += w0 * w0; a1 += w1 * w1; a2 += w2 * w2; a3 += w3 * w3;
        p += 4 * HDIM;
    }
    psq[blockIdx.y * HDIM + h] = (a0 + a1) + (a2 + a3);
}

// ---------------- Kernel 1b: reduce partials -> inv_norm ----------------
__global__ void norm_final(const float* __restrict__ psq, float* __restrict__ inv_norm) {
    int h = blockIdx.x * blockDim.x + threadIdx.x;
    float s = 0.f;
#pragma unroll
    for (int j = 0; j < 8; j++) s += psq[j * HDIM + h];
    inv_norm[h] = rsqrtf(s);
}

// ---------------- Kernel 1c: g f32 -> bf16 (row-major, same layout) ----------------
__global__ void conv_g(const float* __restrict__ G, unsigned short* __restrict__ Gb) {
    size_t i = ((size_t)blockIdx.x * 256 + threadIdx.x) * 8;
    float4 a = *(const float4*)(G + i);
    float4 b = *(const float4*)(G + i + 4);
    uint4 o;
    o.x = pack_bf2(a.x, a.y); o.y = pack_bf2(a.z, a.w);
    o.z = pack_bf2(b.x, b.y); o.w = pack_bf2(b.z, b.w);
    *(uint4*)(Gb + i) = o;
}

// ---------------- Kernel 1d: W f32 [k][h] -> scaled bf16 transposed [h][k] ----------------
__global__ __launch_bounds__(256)
void conv_wt(const float* __restrict__ W, const float* __restrict__ inv_norm,
             unsigned short* __restrict__ Wt) {
    __shared__ __align__(16) unsigned short tile[64 * 72];  // [k][h], padded
    const int h0 = blockIdx.x * 64;
    const int k0 = blockIdx.y * 64;
    const int t  = threadIdx.x;
    const int kr = t >> 4;         // 0..15
    const int hc = (t & 15) * 4;   // 0..60
    float4 inv4 = *(const float4*)(inv_norm + h0 + hc);
#pragma unroll
    for (int i = 0; i < 4; i++) {
        int k = kr + i * 16;
        float4 w = *(const float4*)(W + (size_t)(k0 + k) * HDIM + h0 + hc);
        uint2 s;
        s.x = pack_bf2(w.x * inv4.x, w.y * inv4.y);
        s.y = pack_bf2(w.z * inv4.z, w.w * inv4.w);
        *(uint2*)&tile[k * 72 + hc] = s;
    }
    __syncthreads();
    const int hr = t >> 2;         // 0..63
    const int kc = (t & 3) * 16;   // 0,16,32,48
    unsigned int tmp[16];
#pragma unroll
    for (int j = 0; j < 16; j++) tmp[j] = tile[(kc + j) * 72 + hr];
    uint4 q0, q1;
    q0.x = tmp[0] | (tmp[1] << 16);  q0.y = tmp[2] | (tmp[3] << 16);
    q0.z = tmp[4] | (tmp[5] << 16);  q0.w = tmp[6] | (tmp[7] << 16);
    q1.x = tmp[8] | (tmp[9] << 16);  q1.y = tmp[10] | (tmp[11] << 16);
    q1.z = tmp[12] | (tmp[13] << 16); q1.w = tmp[14] | (tmp[15] << 16);
    unsigned short* o = Wt + (size_t)(h0 + hr) * KDIM + k0 + kc;
    *(uint4*)(o + 0) = q0;
    *(uint4*)(o + 8) = q1;
}

// ---------------- TDM tile load: 128 rows x 32 bf16, row stride KDIM, padded LDS dest ----------------
#if __has_builtin(__builtin_amdgcn_tensor_load_to_lds)
#define HAVE_TDM 1
__device__ __forceinline__ void tdm_load_tile(const unsigned short* gaddr, unsigned lds_off) {
    unsigned long long ga = (unsigned long long)(size_t)gaddr;
    u32x4 g0;
    g0[0] = 1u;                                              // count=1
    g0[1] = lds_off;                                         // lds_addr
    g0[2] = (unsigned)(ga & 0xFFFFFFFFu);                    // global_addr[31:0]
    g0[3] = (unsigned)((ga >> 32) & 0x01FFFFFFu) | (2u << 30);  // addr[56:32], type=2
    i32x8 g1;
    // data_size=1(2B) | pad_enable | pad_interval=3(16dw) | pad_amount=3(4dw)
    g1[0] = (int)((1u << 16) | (1u << 20) | (3u << 22) | (3u << 25));
    g1[1] = (int)(((unsigned)KDIM & 0xFFFFu) << 16);         // tensor_dim0[15:0]
    g1[2] = (int)(((unsigned)KDIM >> 16) & 0xFFFFu);         // tensor_dim0[31:16]; dim1 lo16=0
    g1[3] = (int)(0x10u | (32u << 16));                      // tensor_dim1=1<<20 (hi16); tile_dim0=32
    g1[4] = 128;                                             // tile_dim1=128; tile_dim2=0
    g1[5] = (int)KDIM;                                       // tensor_dim0_stride[31:0]
    g1[6] = 0;
    g1[7] = 0;
    i32x4 z4 = {0, 0, 0, 0};
    i32x8 z8 = {0, 0, 0, 0, 0, 0, 0, 0};
    __builtin_amdgcn_tensor_load_to_lds(g0, g1, z4, z4, z8, 0);  // 6-arg (clang-23) form
}
#else
#define HAVE_TDM 0
#endif

// ---------------- shared epilogue: per-row (max,sumexp) over block's 128 cols ----------------
__device__ __forceinline__ void block_lse_epilogue(
    f32x8 acc[4][2], float* red_max, float* red_sum,
    int t, int lane, int wm, int wh, int mBase, int bx,
    float* __restrict__ pmax, float* __restrict__ psum) {
    const int half = lane >> 4;  // C layout: N=lane&15, vgpr r -> M = r + 8*half
    const int nIdx = lane & 15;
#pragma unroll
    for (int mi = 0; mi < 4; mi++) {
#pragma unroll
        for (int r = 0; r < 8; r++) {
            float v0 = acc[mi][0][r], v1 = acc[mi][1][r];
            float m0 = v0, m1 = v1;
#pragma unroll
            for (int off = 8; off >= 1; off >>= 1) {
                m0 = fmaxf(m0, __shfl_xor(m0, off));
                m1 = fmaxf(m1, __shfl_xor(m1, off));
            }
            float e0 = __expf(v0 - m0), e1 = __expf(v1 - m1);
#pragma unroll
            for (int off = 8; off >= 1; off >>= 1) {
                e0 += __shfl_xor(e0, off);
                e1 += __shfl_xor(e1, off);
            }
            float M = fmaxf(m0, m1);
            float S = __expf(m0 - M) * e0 + __expf(m1 - M) * e1;
            if (nIdx == 0) {
                int row = wm * 64 + mi * 16 + half * 8 + r;
                red_max[row * 4 + wh] = M;
                red_sum[row * 4 + wh] = S;
            }
        }
    }
    __syncthreads();
    if (t < BM) {
        float M = red_max[t * 4 + 0];
        M = fmaxf(M, red_max[t * 4 + 1]);
        M = fmaxf(M, red_max[t * 4 + 2]);
        M = fmaxf(M, red_max[t * 4 + 3]);
        float S = 0.f;
#pragma unroll
        for (int j = 0; j < 4; j++) S += __expf(red_max[t * 4 + j] - M) * red_sum[t * 4 + j];
        size_t o = (size_t)(mBase + t) * HB + bx;
        pmax[o] = M;
        psum[o] = S;
    }
}

// ---------------- Kernel 2 (big path): bf16 WMMA GEMM, TDM double-buffered staging ----------------
__global__ __launch_bounds__(256)
void gemm_lse_tdm(const unsigned short* __restrict__ Gb, const unsigned short* __restrict__ Wt,
                  float* __restrict__ pmax, float* __restrict__ psum) {
    __shared__ __align__(16) unsigned short lds_a[2][BM * LSTR];
    __shared__ __align__(16) unsigned short lds_b[2][BH * LSTR];
    __shared__ float red_max[BM * 4];
    __shared__ float red_sum[BM * 4];

    const int t    = threadIdx.x;
    const int lane = t & 31;
    const int wave = t >> 5;
    const int wm   = wave >> 2;
    const int wh   = wave & 3;
    const int mBase = blockIdx.y * BM;
    const int hBase = blockIdx.x * BH;

    const unsigned short* gA = Gb + (size_t)mBase * KDIM;
    const unsigned short* gB = Wt + (size_t)hBase * KDIM;

    f32x8 acc[4][2];
#pragma unroll
    for (int mi = 0; mi < 4; mi++)
#pragma unroll
        for (int hi = 0; hi < 2; hi++)
#pragma unroll
            for (int e = 0; e < 8; e++) acc[mi][hi][e] = 0.f;

#if HAVE_TDM
    if (wave == 0) {
        tdm_load_tile(gA, (unsigned)(size_t)&lds_a[0][0]);
        tdm_load_tile(gB, (unsigned)(size_t)&lds_b[0][0]);
    }
#endif
    for (int kt = 0; kt < NT; ++kt) {
        const int cur = kt & 1;
#if HAVE_TDM
        if (wave == 0) {
            if (kt + 1 < NT) {
                const int k1 = (kt + 1) * BK;
                tdm_load_tile(gA + k1, (unsigned)(size_t)&lds_a[(kt + 1) & 1][0]);
                tdm_load_tile(gB + k1, (unsigned)(size_t)&lds_b[(kt + 1) & 1][0]);
                __builtin_amdgcn_s_wait_tensorcnt(2);  // 2 newest outstanding -> tile kt complete
            } else {
                __builtin_amdgcn_s_wait_tensorcnt(0);
            }
        }
        __syncthreads();
#else
        // fallback staging: synchronous bf16 copies (no conversion math)
        {
            const int k0  = kt * BK;
            const int row = t >> 1;            // 0..127
            const int seg = (t & 1) * 16;      // ushort offset 0/16
            const uint4* pa = (const uint4*)(gA + (size_t)row * KDIM + k0 + seg);
            const uint4* pb = (const uint4*)(gB + (size_t)row * KDIM + k0 + seg);
            uint4 va0 = pa[0], va1 = pa[1];
            uint4 vb0 = pb[0], vb1 = pb[1];
            __syncthreads();
            *(uint4*)&lds_a[cur][row * LSTR + seg + 0] = va0;
            *(uint4*)&lds_a[cur][row * LSTR + seg + 8] = va1;
            *(uint4*)&lds_b[cur][row * LSTR + seg + 0] = vb0;
            *(uint4*)&lds_b[cur][row * LSTR + seg + 8] = vb1;
            __syncthreads();
        }
#endif
        const int kh = (lane >> 4) * 16;  // same K-half map for A and B -> consistent permutation
        const int fr = lane & 15;
        FragBF a[4], b[2];
#pragma unroll
        for (int mi = 0; mi < 4; mi++) {
            const uint4* p = (const uint4*)&lds_a[cur][(wm * 64 + mi * 16 + fr) * LSTR + kh];
            a[mi].q[0] = p[0]; a[mi].q[1] = p[1];
        }
#pragma unroll
        for (int hi = 0; hi < 2; hi++) {
            const uint4* p = (const uint4*)&lds_b[cur][(wh * 32 + hi * 16 + fr) * LSTR + kh];
            b[hi].q[0] = p[0]; b[hi].q[1] = p[1];
        }
#pragma unroll
        for (int mi = 0; mi < 4; mi++)
#pragma unroll
            for (int hi = 0; hi < 2; hi++)
                acc[mi][hi] = __builtin_amdgcn_wmma_f32_16x16x32_bf16(
                    false, a[mi].v, false, b[hi].v, (short)0, acc[mi][hi], false, false);
#if HAVE_TDM
        __syncthreads();  // all reads of buf done before wave0 re-targets it next iter
#endif
    }
    block_lse_epilogue(acc, red_max, red_sum, t, lane, wm, wh, mBase, blockIdx.x, pmax, psum);
}

// ---------------- Kernel 2 (fallback path, small ws): fused f32->bf16 GEMM ----------------
__global__ __launch_bounds__(256)
void gemm_lse_fused(const float* __restrict__ G, const float* __restrict__ W,
                    const float* __restrict__ inv_norm,
                    float* __restrict__ pmax, float* __restrict__ psum) {
    __shared__ __align__(16) unsigned short lds_a[BM * LSTR];
    __shared__ __align__(16) unsigned short lds_b[BH * LSTR];
    __shared__ float red_max[BM * 4];
    __shared__ float red_sum[BM * 4];

    const int t    = threadIdx.x;
    const int lane = t & 31;
    const int wave = t >> 5;
    const int wm   = wave >> 2;
    const int wh   = wave & 3;
    const int mBase = blockIdx.y * BM;
    const int hBase = blockIdx.x * BH;

    const int arow = t >> 3;
    const int atk  = (t & 7) * 4;
    const int hcol = (t & 63) * 2;
    const int kr0  = (t >> 6) * 8;

    const float2 invp = *(const float2*)(inv_norm + hBase + hcol);

    f32x8 acc[4][2];
#pragma unroll
    for (int mi = 0; mi < 4; mi++)
#pragma unroll
        for (int hi = 0; hi < 2; hi++)
#pragma unroll
            for (int e = 0; e < 8; e++) acc[mi][hi][e] = 0.f;

    for (int kt = 0; kt < NT; ++kt) {
        const int k0 = kt * BK;
        float4 ga[4];
#pragma unroll
        for (int i = 0; i < 4; i++)
            ga[i] = *(const float4*)(G + (size_t)(mBase + arow + i * 32) * KDIM + k0 + atk);
        float2 gw[8];
#pragma unroll
        for (int j = 0; j < 8; j++)
            gw[j] = *(const float2*)(W + (size_t)(k0 + kr0 + j) * HDIM + hBase + hcol);

        if (kt + 1 < NT) {
            __builtin_prefetch(G + (size_t)(mBase + arow) * KDIM + k0 + BK + atk, 0, 0);
            __builtin_prefetch(W + (size_t)(k0 + BK + kr0) * HDIM + hBase + hcol, 0, 0);
        }

        __syncthreads();
#pragma unroll
        for (int i = 0; i < 4; i++) {
            uint2 s; s.x = pack_bf2(ga[i].x, ga[i].y); s.y = pack_bf2(ga[i].z, ga[i].w);
            *(uint2*)&lds_a[(arow + i * 32) * LSTR + atk] = s;
        }
        uint4 c0, c1;
        c0.x = pack_bf2(gw[0].x * invp.x, gw[1].x * invp.x);
        c0.y = pack_bf2(gw[2].x * invp.x, gw[3].x * invp.x);
        c0.z = pack_bf2(gw[4].x * invp.x, gw[5].x * invp.x);
        c0.w = pack_bf2(gw[6].x * invp.x, gw[7].x * invp.x);
        c1.x = pack_bf2(gw[0].y * invp.y, gw[1].y * invp.y);
        c1.y = pack_bf2(gw[2].y * invp.y, gw[3].y * invp.y);
        c1.z = pack_bf2(gw[4].y * invp.y, gw[5].y * invp.y);
        c1.w = pack_bf2(gw[6].y * invp.y, gw[7].y * invp.y);
        *(uint4*)&lds_b[(hcol + 0) * LSTR + kr0] = c0;
        *(uint4*)&lds_b[(hcol + 1) * LSTR + kr0] = c1;
        __syncthreads();

        const int kh = (lane >> 4) * 16;
        const int fr = lane & 15;
        FragBF a[4], b[2];
#pragma unroll
        for (int mi = 0; mi < 4; mi++) {
            const uint4* p = (const uint4*)&lds_a[(wm * 64 + mi * 16 + fr) * LSTR + kh];
            a[mi].q[0] = p[0]; a[mi].q[1] = p[1];
        }
#pragma unroll
        for (int hi = 0; hi < 2; hi++) {
            const uint4* p = (const uint4*)&lds_b[(wh * 32 + hi * 16 + fr) * LSTR + kh];
            b[hi].q[0] = p[0]; b[hi].q[1] = p[1];
        }
#pragma unroll
        for (int mi = 0; mi < 4; mi++)
#pragma unroll
            for (int hi = 0; hi < 2; hi++)
                acc[mi][hi] = __builtin_amdgcn_wmma_f32_16x16x32_bf16(
                    false, a[mi].v, false, b[hi].v, (short)0, acc[mi][hi], false, false);
    }
    block_lse_epilogue(acc, red_max, red_sum, t, lane, wm, wh, mBase, blockIdx.x, pmax, psum);
}

// ---------------- Kernel 3: combine logsumexp partials per row ----------------
__global__ void lse_final(const float* __restrict__ pmax, const float* __restrict__ psum,
                          float* __restrict__ out) {
    __shared__ float sm[HB];
    const int row = blockIdx.x;
    const int t   = threadIdx.x;
    float m = pmax[(size_t)row * HB + t];
    sm[t] = m;
    __syncthreads();
    for (int s = HB / 2; s > 0; s >>= 1) {
        if (t < s) sm[t] = fmaxf(sm[t], sm[t + s]);
        __syncthreads();
    }
    float gmax = sm[0];
    __syncthreads();
    sm[t] = __expf(m - gmax) * psum[(size_t)row * HB + t];
    __syncthreads();
    for (int s = HB / 2; s > 0; s >>= 1) {
        if (t < s) sm[t] += sm[t + s];
        __syncthreads();
    }
    if (t == 0) out[row] = gmax + logf(sm[0]);
}

extern "C" void kernel_launch(void* const* d_in, const int* in_sizes, int n_in,
                              void* d_out, int out_size, void* d_ws, size_t ws_size,
                              hipStream_t stream) {
    const float* G = (const float*)d_in[0];  // [4,2048,4096] f32
    const float* W = (const float*)d_in[1];  // [4096,16384] f32
    char* wsb = (char*)d_ws;
    float* inv_norm = (float*)wsb;                     // 16384 f
    float* pmax     = inv_norm + HDIM;                 // MDIM*HB f
    float* psum     = pmax + (size_t)MDIM * HB;        // MDIM*HB f
    float* psq      = psum + (size_t)MDIM * HB;        // 8*HDIM f
    unsigned short* Gb = (unsigned short*)(psq + 8 * HDIM);   // MDIM*KDIM bf16
    unsigned short* Wt = Gb + (size_t)MDIM * KDIM;            // HDIM*KDIM bf16
    size_t need_big = (size_t)((char*)(Wt + (size_t)HDIM * KDIM) - wsb);

    norm_partial<<<dim3(HDIM / 256, 8), 256, 0, stream>>>(W, psq);
    norm_final<<<HDIM / 256, 256, 0, stream>>>(psq, inv_norm);
    if (ws_size >= need_big) {
        conv_g<<<(MDIM * (size_t)KDIM) / 8 / 256, 256, 0, stream>>>(G, Gb);
        conv_wt<<<dim3(HDIM / 64, KDIM / 64), 256, 0, stream>>>(W, inv_norm, Wt);
        gemm_lse_tdm<<<dim3(HDIM / BH, MDIM / BM), 256, 0, stream>>>(Gb, Wt, pmax, psum);
    } else {
        gemm_lse_fused<<<dim3(HDIM / BH, MDIM / BM), 256, 0, stream>>>(G, W, inv_norm, pmax, psum);
    }
    lse_final<<<MDIM, HB, 0, stream>>>(pmax, psum, (float*)d_out);
}
